// SpatialAttention_20452634263750
// MI455X (gfx1250) — matrix operand order repaired and checked by reading, not compiled, not run on hardware
//
#include <hip/hip_runtime.h>
#include <hip/hip_fp16.h>

typedef __attribute__((ext_vector_type(16))) _Float16 v16h;
typedef __attribute__((ext_vector_type(8)))  _Float16 v8h;
typedef __attribute__((ext_vector_type(8)))  float    v8f;

#define L2EPS 1e-12f

// Load a 16-half WMMA fragment as two aligned 16-byte chunks: halves [0..7] at p, [8..15] at p+16.
__device__ __forceinline__ v16h load_frag16(const _Float16* p) {
    v8h lo = *(const v8h*)(p);
    v8h hi = *(const v8h*)(p + 16);
    return __builtin_shufflevector(lo, hi, 0,1,2,3,4,5,6,7,8,9,10,11,12,13,14,15);
}

// ---------------------------------------------------------------------------
// Q projection + row L2 norm.  q[b,d,n,h,w] = sum_c x[b,c,n,h,w]*w_q[d,c]
// Store uq[b][p=h*64+w][f=d*8+n] as f16, row-normalized over f (256).
// grid (256, 8), block 256; each block: 16 consecutive w positions.
// ---------------------------------------------------------------------------
__global__ void qproj_kernel(const float* __restrict__ x,
                             const float* __restrict__ w_q,
                             _Float16* __restrict__ uq) {
    __shared__ float xs[64 * 8 * 16];   // [(c*8+nn)*16 + j]   32KB
    __shared__ float wq[32 * 64];       // 8KB
    __shared__ float part[256];
    __shared__ float rinv[16];

    const int t  = threadIdx.x;
    const int b  = blockIdx.y;
    const int p0 = blockIdx.x * 16;

    const float* xb = x + (size_t)b * 64 * 8 * 4096 + p0;
    for (int l = t; l < 64 * 8 * 16; l += 256) {
        int cn = l >> 4, j = l & 15;
        xs[l] = xb[(size_t)cn * 4096 + j];
    }
    for (int l = t; l < 32 * 64; l += 256) wq[l] = w_q[l];
    __syncthreads();

    const int d = t >> 3, nn = t & 7;   // f = t
    float acc[16];
#pragma unroll
    for (int j = 0; j < 16; ++j) acc[j] = 0.f;
    for (int c = 0; c < 64; ++c) {
        float wv = wq[d * 64 + c];
        const float* xr = &xs[(c * 8 + nn) * 16];
#pragma unroll
        for (int j = 0; j < 16; ++j) acc[j] += xr[j] * wv;
    }
    __syncthreads();                      // done with xs; reuse as q buffer
    float (*qs)[256] = (float(*)[256])xs; // qs[pos][f]
#pragma unroll
    for (int j = 0; j < 16; ++j) qs[j][t] = acc[j];
    __syncthreads();

    {   // per-position sum of squares over f
        int pos = t >> 4, ch = t & 15;
        float s = 0.f;
#pragma unroll
        for (int i = 0; i < 16; ++i) { float v = qs[pos][ch * 16 + i]; s += v * v; }
        part[t] = s;
    }
    __syncthreads();
    if (t < 16) {
        float s = 0.f;
#pragma unroll
        for (int ch = 0; ch < 16; ++ch) s += part[t * 16 + ch];
        rinv[t] = 1.0f / fmaxf(sqrtf(s), L2EPS);
    }
    __syncthreads();

    _Float16* uqb = uq + ((size_t)b * 4096 + p0) * 256 + t;
#pragma unroll
    for (int j = 0; j < 16; ++j)
        uqb[(size_t)j * 256] = (_Float16)(acc[j] * rinv[j]);
}

// ---------------------------------------------------------------------------
// KV projection on x[:, :, :, ::2, ::2].  kv[e] = sum_c x_s*w_kv[e,c].
// k = kv[:32] column-normalized over f; v = kv[32:].
// ukT[b][p2][f] (f16, key-major), uvT[b][f][p2] (f16).
// grid (64, 8), block 256; block covers 16 consecutive w2.
// ---------------------------------------------------------------------------
__global__ void kvproj_kernel(const float* __restrict__ x,
                              const float* __restrict__ w_kv,
                              _Float16* __restrict__ ukT,
                              _Float16* __restrict__ uvT) {
    __shared__ float xs[64 * 8 * 16];   // 32KB
    __shared__ float wk[64 * 64];       // 16KB
    __shared__ float part[256];
    __shared__ float rinv[16];

    const int t   = threadIdx.x;
    const int b   = blockIdx.y;
    const int p20 = blockIdx.x * 16;    // p2 = h2*32 + w2
    const int h2  = p20 >> 5;
    const int w20 = p20 & 31;

    const float* xb = x + (size_t)b * 64 * 8 * 4096 + (size_t)h2 * 128 + (size_t)w20 * 2;
    for (int l = t; l < 64 * 8 * 16; l += 256) {
        int cn = l >> 4, j = l & 15;
        xs[l] = xb[(size_t)cn * 4096 + j * 2];
    }
    for (int l = t; l < 64 * 64; l += 256) wk[l] = w_kv[l];
    __syncthreads();

    const int d = t >> 3, nn = t & 7;   // f = t
    float ka[16], va[16];
#pragma unroll
    for (int j = 0; j < 16; ++j) { ka[j] = 0.f; va[j] = 0.f; }
    for (int c = 0; c < 64; ++c) {
        float wkk = wk[d * 64 + c];
        float wvv = wk[(d + 32) * 64 + c];
        const float* xr = &xs[(c * 8 + nn) * 16];
#pragma unroll
        for (int j = 0; j < 16; ++j) { ka[j] += xr[j] * wkk; va[j] += xr[j] * wvv; }
    }
    __syncthreads();
    float (*ks)[256] = (float(*)[256])xs;
#pragma unroll
    for (int j = 0; j < 16; ++j) ks[j][t] = ka[j];
    __syncthreads();

    {
        int pos = t >> 4, ch = t & 15;
        float s = 0.f;
#pragma unroll
        for (int i = 0; i < 16; ++i) { float v = ks[pos][ch * 16 + i]; s += v * v; }
        part[t] = s;
    }
    __syncthreads();
    if (t < 16) {
        float s = 0.f;
#pragma unroll
        for (int ch = 0; ch < 16; ++ch) s += part[t * 16 + ch];
        rinv[t] = 1.0f / fmaxf(sqrtf(s), L2EPS);
    }
    __syncthreads();

    _Float16* ukb = ukT + ((size_t)b * 1024 + p20) * 256 + t;
    _Float16* uvb = uvT + ((size_t)b * 256 + t) * 1024 + p20;
#pragma unroll
    for (int j = 0; j < 16; ++j) {
        ukb[(size_t)j * 256] = (_Float16)(ka[j] * rinv[j]);
        uvb[j]               = (_Float16)va[j];
    }
}

// ---------------------------------------------------------------------------
// Attention: a wave QUAD handles one 16-query tile; 64 keys per iteration.
//   sub = wave&3: QK -> S^T sub-tile for keys [sub*16, sub*16+16) of each
//   64-key chunk (8 WMMAs), exp(), publish to LDS.  After barrier each wave
//   rebuilds the two shared K=32 P B-fragments from LDS and runs PV for its
//   4 output-feature tiles (8 WMMAs).  oacc = 32 VGPRs, Q fragments = 64
//   VGPRs preloaded once.  __launch_bounds__(256,4) raises the VGPR cap
//   (~256) so the whole working set stays in registers -> no scratch spills.
//   Scores <= ~1 (unit rows/cols) so exp() needs no max subtraction.
// grid (128, 8), block 256 (8 waves = 2 quads -> 2 query tiles / block).
// ---------------------------------------------------------------------------
__global__ void __launch_bounds__(256, 4)
attn_kernel(const _Float16* __restrict__ uq,
            const _Float16* __restrict__ ukT,
            const _Float16* __restrict__ uvT,
            float* __restrict__ out) {
    __shared__ float pbuf[2][4][16][16]; // [quad][sub-tile][key row][qcol]  8KB
    __shared__ float sbuf[2][4][16];     // per-quad partial softmax sums

    const int lane = threadIdx.x & 31;
    const int wid  = threadIdx.x >> 5;
    const int quad = wid >> 2;           // 0..1
    const int sub  = wid & 3;            // 16-key sub-tile / 4 ft tiles
    const int b    = blockIdx.y;
    const int qtile = blockIdx.x * 2 + quad;
    const int col  = lane & 15;          // query column inside tile
    const int hi   = lane >> 4;
    const int qrow = qtile * 16 + col;

    const _Float16* uqp = uq  + ((size_t)b * 4096 + qrow) * 256;
    const _Float16* ukp = ukT + (size_t)b * 1024 * 256;
    const _Float16* uvp = uvT + (size_t)b * 256 * 1024;

    // Preload all 8 Q B-fragments (64 VGPRs), reused for every key chunk.
    v16h qf[8];
#pragma unroll
    for (int fc = 0; fc < 8; ++fc)
        qf[fc] = load_frag16(uqp + fc * 32 + hi * 8);

    v8f oacc[4];
#pragma unroll
    for (int ft = 0; ft < 4; ++ft)
        oacc[ft] = (v8f){0.f,0.f,0.f,0.f,0.f,0.f,0.f,0.f};
    float ssum = 0.f;

    for (int kc = 0; kc < 16; ++kc) {          // 16 chunks of 64 keys
        const int key0 = kc * 64;

        // ---- S^T sub-tile: this wave's 16 keys, K-dim = feature (256) ----
        v8f s = (v8f){0.f,0.f,0.f,0.f,0.f,0.f,0.f,0.f};
        const _Float16* arow = ukp + (size_t)(key0 + sub * 16 + col) * 256 + hi * 8;
#pragma unroll
        for (int fc = 0; fc < 8; ++fc) {
            v16h a = load_frag16(arow + fc * 32);
            s = __builtin_amdgcn_wmma_f32_16x16x32_f16(false, a, false, qf[fc], (short)0, s, false, false);
        }

        // ---- exp, local softmax-sum, publish sub-tile to LDS ----
#pragma unroll
        for (int r = 0; r < 8; ++r) {
            float e = __expf(s[r]);
            ssum += e;
            pbuf[quad][sub][hi * 8 + r][col] = e;
        }
        __syncthreads();

        // ---- PV: two K=32 P fragments cover this 64-key chunk ----
#pragma unroll
        for (int half = 0; half < 2; ++half) {
            // B layout: halves[m] = P[key = half*32 + (m>>3)*16 + hi*8 + (m&7)][qcol]
            v16h pf;
            const float* src = &pbuf[quad][half * 2][hi * 8][col];
#pragma unroll
            for (int m = 0; m < 16; ++m)
                pf[m] = (_Float16)src[(m >> 3) * 256 + (m & 7) * 16];

#pragma unroll
            for (int ft = 0; ft < 4; ++ft) {
                const int ftg = sub * 4 + ft;
                v16h af = load_frag16(uvp + (size_t)(ftg * 16 + col) * 1024
                                          + key0 + half * 32 + hi * 8);
                oacc[ft] = __builtin_amdgcn_wmma_f32_16x16x32_f16(false, af, false, pf, (short)0, oacc[ft], false, false);
            }
        }
        __syncthreads();                 // before pbuf is overwritten
    }

    // ---- finish softmax denominator across the quad ----
    const float part_tot = ssum + __shfl_xor(ssum, 16, 32);
    sbuf[quad][sub][col] = part_tot;     // lanes l and l+16 write same value
    __syncthreads();
    const float inv = 1.0f / (sbuf[quad][0][col] + sbuf[quad][1][col] +
                              sbuf[quad][2][col] + sbuf[quad][3][col]);

    float* outp = out + (size_t)b * 256 * 4096 + qtile * 16 + col;
#pragma unroll
    for (int ft = 0; ft < 4; ++ft)
#pragma unroll
        for (int r = 0; r < 8; ++r) {
            int f = (sub * 4 + ft) * 16 + hi * 8 + r;
            outp[(size_t)f * 4096] = oacc[ft][r] * inv;
        }
}

extern "C" void kernel_launch(void* const* d_in, const int* in_sizes, int n_in,
                              void* d_out, int out_size, void* d_ws, size_t ws_size,
                              hipStream_t stream) {
    const float* x    = (const float*)d_in[0];   // (8,64,8,64,64)
    const float* w_q  = (const float*)d_in[1];   // (32,64)
    const float* w_kv = (const float*)d_in[2];   // (64,64)
    float* out = (float*)d_out;                  // (8,32,8,64,64) == OutT[b][f][p]

    _Float16* uq  = (_Float16*)d_ws;             // [8][4096][256]  16MB
    _Float16* ukT = uq  + (size_t)8 * 4096 * 256;// [8][1024][256]   4MB
    _Float16* uvT = ukT + (size_t)8 * 1024 * 256;// [8][256][1024]   4MB

    qproj_kernel <<<dim3(256, 8), 256, 0, stream>>>(x, w_q, uq);
    kvproj_kernel<<<dim3(64, 8),  256, 0, stream>>>(x, w_kv, ukT, uvT);
    attn_kernel  <<<dim3(128, 8), 256, 0, stream>>>(uq, ukT, uvT, out);
}
